// LSTMClassifier_29661044146580
// MI455X (gfx1250) — matrix operand order repaired and checked by reading, not compiled, not run on hardware
//
#include <hip/hip_runtime.h>
#include <hip/hip_bf16.h>
#include <math.h>

// ---------------------------------------------------------------------------
// LSTM classifier forward for MI455X (gfx1250, wave32, WMMA).
//   B=128, T=256, EMB=300(->320), HID=1024, 4H=4096, OUT=3, PAD=1
// Per step, one fused kernel computes for its 16 hidden columns j0..j0+15:
//   pre[g][m][j] = bsum + [h | emb_t] @ [W_hh | W_ih]^T   (all 4 gates)
// via v_wmma_f32_16x16x32_bf16 (K = 1344 = 42 chunks of 32), then applies the
// LSTM cell (sigmoid/tanh, c/h update, final-hidden capture) in registers.
// Global->LDS staging uses CDNA5 async-to-LDS ops (ASYNCcnt) when available,
// double-buffered so the copy of chunk k+1 overlaps WMMA on chunk k.
// ---------------------------------------------------------------------------

typedef __attribute__((ext_vector_type(16))) __bf16 v16bf;
typedef __attribute__((ext_vector_type(8)))  __bf16 v8bf;
typedef __attribute__((ext_vector_type(8)))  float  v8f;

#define BATCH   128
#define SEQT    256
#define EMB     300
#define EMBP    320
#define HID     1024
#define G4      4096
#define OUTN    3
#define PAD_IDX 1
#define KTOT    (HID + EMBP)   // 1344
#define KCHUNKS (KTOT / 32)    // 42
#define LDSTR   40             // LDS row stride (32 + 8 pad) in bf16 elems

// ---- CDNA5 async global->LDS (guarded: falls back to sync copy) ----
#if defined(__HIP_DEVICE_COMPILE__) && __has_builtin(__builtin_amdgcn_global_load_async_to_lds_b128)
#define ASYNC_LDS 1
typedef int v4i_t __attribute__((vector_size(16)));
typedef __attribute__((address_space(1))) v4i_t* gv4i_p;   // global (device) AS
typedef __attribute__((address_space(3))) v4i_t* lv4i_p;   // LDS AS
static __device__ __forceinline__ void async_cp16(const void* g, void* l) {
  __builtin_amdgcn_global_load_async_to_lds_b128((gv4i_p)(g), (lv4i_p)(l), 0, 0);
}
static __device__ __forceinline__ void async_wait0() {
#if __has_builtin(__builtin_amdgcn_s_wait_asynccnt)
  __builtin_amdgcn_s_wait_asynccnt(0);
#else
  asm volatile("s_wait_asynccnt 0x0" ::: "memory");
#endif
}
#else
#define ASYNC_LDS 0
static __device__ __forceinline__ void async_wait0() {}
#endif

// float -> bf16, round-to-nearest-even (storage-only)
static __device__ __forceinline__ __bf16 f2bf(float f) {
  unsigned u = __builtin_bit_cast(unsigned, f);
  unsigned r = u + 0x7FFFu + ((u >> 16) & 1u);
  return __builtin_bit_cast(__bf16, (unsigned short)(r >> 16));
}

static __device__ __forceinline__ float sigmoidf_(float x) {
  return 1.0f / (1.0f + expf(-x));
}

// ---------------- prep kernels (run once per launch) ----------------

__global__ void k_prep_whh(const float* __restrict__ whh, __bf16* __restrict__ out) {
  int i = blockIdx.x * blockDim.x + threadIdx.x;
  if (i < G4 * HID) out[i] = f2bf(whh[i]);
}

__global__ void k_prep_wih(const float* __restrict__ wih, __bf16* __restrict__ out) {
  int i = blockIdx.x * blockDim.x + threadIdx.x;
  if (i < G4 * EMBP) {
    int g = i / EMBP, e = i % EMBP;
    out[i] = (e < EMB) ? f2bf(wih[g * EMB + e]) : f2bf(0.0f);
  }
}

__global__ void k_prep_bias_state(const float* __restrict__ b_ih,
                                  const float* __restrict__ b_hh,
                                  float* __restrict__ bsum,
                                  __bf16* __restrict__ h0,
                                  float* __restrict__ c,
                                  float* __restrict__ fin) {
  int i = blockIdx.x * blockDim.x + threadIdx.x;
  if (i < G4) bsum[i] = b_ih[i] + b_hh[i];
  if (i < BATCH * HID) { h0[i] = f2bf(0.0f); c[i] = 0.0f; fin[i] = 0.0f; }
}

__global__ void k_lengths(const int* __restrict__ x, int* __restrict__ len) {
  int b = threadIdx.x;  // blockDim == 128
  int n = 0;
  for (int t = 0; t < SEQT; ++t) n += (x[b * SEQT + t] != PAD_IDX) ? 1 : 0;
  len[b] = n;
}

// one block per token; 320 threads; pad row of embed is already zero
__global__ void k_embed(const int* __restrict__ x, const float* __restrict__ embed,
                        __bf16* __restrict__ inpB) {
  int row = blockIdx.x;       // b*T + t
  int e = threadIdx.x;        // 0..319
  int tok = x[row];
  float v = (e < EMB) ? embed[tok * EMB + e] : 0.0f;
  inpB[row * EMBP + e] = f2bf(v);
}

// ---------------- fused recurrent step: GEMM (WMMA) + LSTM cell ----------------
// grid = (HID/16, BATCH/64) = (64, 2); block = 128 threads = 4 waves.
// Block owns hidden columns j0..j0+15 for all 4 gates and batch rows m0..m0+63.
// Wave w -> M strip [m0+16w, +16); acc[ns] = gate ns pre-activations.
// B-tile row r maps to weight row (r>>4)*HID + j0 + (r&15)  (gate-major).

__global__ __launch_bounds__(128) void k_step(
    const __bf16* __restrict__ hin,    // [BATCH][HID] bf16 (step t input)
    __bf16* __restrict__ hout,         // [BATCH][HID] bf16 (step t output)
    const __bf16* __restrict__ inpB,   // [BATCH*SEQT][EMBP]
    const __bf16* __restrict__ whh,    // [G4][HID]
    const __bf16* __restrict__ wih,    // [G4][EMBP]
    const float*  __restrict__ bsum,   // [G4] b_ih + b_hh
    float* __restrict__ cbuf,          // [BATCH][HID]
    float* __restrict__ fin,           // [BATCH][HID] captured last hidden
    const int* __restrict__ x,         // [BATCH][SEQT]
    const int* __restrict__ len,       // [BATCH]
    int t) {
  __shared__ __align__(16) __bf16 lA[2][64 * LDSTR];
  __shared__ __align__(16) __bf16 lB[2][64 * LDSTR];

  const int tid  = threadIdx.x;
  const int wave = tid >> 5;
  const int lane = tid & 31;
  const int lh   = lane >> 4;   // 0: lanes 0-15, 1: lanes 16-31
  const int ll   = lane & 15;
  const int j0   = blockIdx.x * 16;   // hidden column base (per gate)
  const int m0   = blockIdx.y * 64;   // batch row base

  // accumulators pre-seeded with bias (bias independent of m)
  v8f acc[4];
  #pragma unroll
  for (int ns = 0; ns < 4; ++ns) {
    float bv = bsum[ns * HID + j0 + ll];
    #pragma unroll
    for (int i = 0; i < 8; ++i) acc[ns][i] = bv;
  }

  // cooperative tile copy: A(64x32 of [h|emb_t]) and B(64x32 of gate-major W)
  auto load_tiles = [&](int kc, int buf) {
    const int kk = kc * 32;
    #pragma unroll
    for (int cpass = 0; cpass < 2; ++cpass) {
      int ch  = tid + cpass * 128;
      int r   = ch >> 2;        // tile row
      int sub = ch & 3;         // 16B sub-chunk
      int wrow = (r >> 4) * HID + j0 + (r & 15);   // weight row (gate-major)
      const __bf16* srcA;
      const __bf16* srcB;
      if (kk < HID) {
        srcA = hin + (m0 + r) * HID + kk + sub * 8;
        srcB = whh + wrow * HID + kk + sub * 8;
      } else {
        srcA = inpB + ((m0 + r) * SEQT + t) * EMBP + (kk - HID) + sub * 8;
        srcB = wih  + wrow * EMBP + (kk - HID) + sub * 8;
      }
#if ASYNC_LDS
      async_cp16(srcA, &lA[buf][r * LDSTR + sub * 8]);
      async_cp16(srcB, &lB[buf][r * LDSTR + sub * 8]);
#else
      __builtin_prefetch(srcA + 32, 0, 1);
      __builtin_prefetch(srcB + 32, 0, 1);
      *(uint4*)&lA[buf][r * LDSTR + sub * 8] = *(const uint4*)srcA;
      *(uint4*)&lB[buf][r * LDSTR + sub * 8] = *(const uint4*)srcB;
#endif
    }
  };

  load_tiles(0, 0);
  async_wait0();
  __syncthreads();

  for (int kc = 0; kc < KCHUNKS; ++kc) {
    const int cur = kc & 1;
    if (kc + 1 < KCHUNKS) load_tiles(kc + 1, cur ^ 1);  // overlaps compute below

    // A fragment (ISA 16-bit A 16x32 layout):
    // lane<16: K {0..7}+{16..23}; lane>=16: K {8..15}+{24..31}; row = ll
    const int ar = wave * 16 + ll;
    v8bf alo = *(const v8bf*)&lA[cur][ar * LDSTR + lh * 8];
    v8bf ahi = *(const v8bf*)&lA[cur][ar * LDSTR + 16 + lh * 8];
    v16bf a;
    #pragma unroll
    for (int i = 0; i < 8; ++i) { a[i] = alo[i]; a[i + 8] = ahi[i]; }

    // B fragments: lane holds column N=ll of gate ns, K {lh*8..+7}+{16+lh*8..+7}
    #pragma unroll
    for (int ns = 0; ns < 4; ++ns) {
      const int br = ns * 16 + ll;
      v8bf blo = *(const v8bf*)&lB[cur][br * LDSTR + lh * 8];
      v8bf bhi = *(const v8bf*)&lB[cur][br * LDSTR + 16 + lh * 8];
      v16bf bb;
      #pragma unroll
      for (int i = 0; i < 8; ++i) { bb[i] = blo[i]; bb[i + 8] = bhi[i]; }
      acc[ns] = __builtin_amdgcn_wmma_f32_16x16x32_bf16(
          false, a, false, bb, (short)0, acc[ns], false, false);
    }

    async_wait0();    // next-chunk async copies landed in LDS
    __syncthreads();  // everyone done reading cur / writing nxt
  }

  // ---- fused LSTM cell: i,f,g,o for (m, j) are acc[0..3][i] in this lane ----
  const int mbase = m0 + wave * 16 + lh * 8;   // C/D layout: VGPR i -> M=i(+8)
  const int j = j0 + ll;
  #pragma unroll
  for (int i = 0; i < 8; ++i) {
    const int m = mbase + i;
    float gi = sigmoidf_(acc[0][i]);
    float gf = sigmoidf_(acc[1][i]);
    float gg = tanhf(acc[2][i]);
    float go = sigmoidf_(acc[3][i]);
    const int cidx = m * HID + j;
    float cn = gf * cbuf[cidx] + gi * gg;
    float hn = go * tanhf(cn);
    cbuf[cidx] = cn;
    hout[cidx] = f2bf(hn);
    if (t == len[m] - 1) {
      int tok = x[m * SEQT + t];
      fin[cidx] = (tok == PAD_IDX) ? 0.0f : hn;  // reference masks pads in outs
    }
  }
}

// ---------------- output projection [128,3] ----------------

__global__ void k_out(const float* __restrict__ fin,
                      const float* __restrict__ W_out,
                      const float* __restrict__ b_out,
                      float* __restrict__ out) {
  int idx = blockIdx.x * blockDim.x + threadIdx.x;
  if (idx >= BATCH * OUTN) return;
  int b = idx / OUTN, o = idx % OUTN;
  float s = b_out[o];
  for (int k = 0; k < HID; ++k) s += fin[b * HID + k] * W_out[o * HID + k];
  out[b * OUTN + o] = s;
}

// ---------------- workspace layout (bytes, 256-aligned) ----------------
#define OFF_WHH   ((size_t)0)            //  8,388,608  W_hh bf16
#define OFF_WIH   ((size_t)8388608)      //  2,621,440  W_ih bf16 padded
#define OFF_INPB  ((size_t)11010048)     // 20,971,520  embedded tokens bf16
#define OFF_H0    ((size_t)31981568)     //    262,144  h ping
#define OFF_H1    ((size_t)32243712)     //    262,144  h pong
#define OFF_C     ((size_t)32505856)     //    524,288  c f32
#define OFF_FIN   ((size_t)33030144)     //    524,288  final hidden f32
#define OFF_BSUM  ((size_t)33554432)     //     16,384  b_ih+b_hh
#define OFF_LEN   ((size_t)33570816)     //        512  lengths
// total ~33.6 MB

extern "C" void kernel_launch(void* const* d_in, const int* in_sizes, int n_in,
                              void* d_out, int out_size, void* d_ws, size_t ws_size,
                              hipStream_t stream) {
  const int*   x     = (const int*)  d_in[0];
  const float* embed = (const float*)d_in[1];
  const float* W_ih  = (const float*)d_in[2];
  const float* W_hh  = (const float*)d_in[3];
  const float* b_ih  = (const float*)d_in[4];
  const float* b_hh  = (const float*)d_in[5];
  const float* W_out = (const float*)d_in[6];
  const float* b_out = (const float*)d_in[7];
  float* out = (float*)d_out;

  char* ws = (char*)d_ws;
  __bf16* whhb = (__bf16*)(ws + OFF_WHH);
  __bf16* wihb = (__bf16*)(ws + OFF_WIH);
  __bf16* inpB = (__bf16*)(ws + OFF_INPB);
  __bf16* h0   = (__bf16*)(ws + OFF_H0);
  __bf16* h1   = (__bf16*)(ws + OFF_H1);
  float*  cbuf = (float*) (ws + OFF_C);
  float*  finb = (float*) (ws + OFF_FIN);
  float*  bsum = (float*) (ws + OFF_BSUM);
  int*    len  = (int*)   (ws + OFF_LEN);

  // per-launch prep (deterministic; redone every call)
  k_prep_whh<<<(G4 * HID + 255) / 256, 256, 0, stream>>>(W_hh, whhb);
  k_prep_wih<<<(G4 * EMBP + 255) / 256, 256, 0, stream>>>(W_ih, wihb);
  k_prep_bias_state<<<(BATCH * HID + 255) / 256, 256, 0, stream>>>(
      b_ih, b_hh, bsum, h0, cbuf, finb);
  k_lengths<<<1, BATCH, 0, stream>>>(x, len);
  k_embed<<<BATCH * SEQT, EMBP, 0, stream>>>(x, embed, inpB);

  // sequential recurrence: 256 fused WMMA-GEMM + cell kernels, h ping-pongs
  dim3 gg(HID / 16, BATCH / 64);   // (64, 2)
  __bf16* hbufs[2] = {h0, h1};
  for (int t = 0; t < SEQT; ++t) {
    k_step<<<gg, 128, 0, stream>>>(hbufs[t & 1], hbufs[(t + 1) & 1], inpB,
                                   whhb, wihb, bsum, cbuf, finb, x, len, t);
  }

  k_out<<<(BATCH * OUTN + 127) / 128, 128, 0, stream>>>(finb, W_out, b_out, out);
}
